// RGAT_18047452578192
// MI455X (gfx1250) — matrix operand order repaired and checked by reading, not compiled, not run on hardware
//
#include <hip/hip_runtime.h>
#include <cstdint>

#define Nn 100000
#define Ee 1600000
#define ET (Ee + Nn)   // edges + self loops

typedef __attribute__((ext_vector_type(16))) _Float16 v16h;
typedef __attribute__((ext_vector_type(8)))  float    v8f;

// ---------------- helpers ----------------
__device__ __forceinline__ float wave_sum(float v) {
#pragma unroll
  for (int off = 16; off > 0; off >>= 1) v += __shfl_xor(v, off, 32);
  return v;
}
__device__ __forceinline__ float wave_max(float v) {
#pragma unroll
  for (int off = 16; off > 0; off >>= 1) v = fmaxf(v, __shfl_xor(v, off, 32));
  return v;
}

__global__ void fill_f32(float* __restrict__ p, long n, float v) {
  long i = (long)blockIdx.x * blockDim.x + threadIdx.x;
  if (i < n) p[i] = v;
}

// ---------------- per-node L2 norm (wave per node, 128 feats) ----------------
__global__ void node_norm(const float* __restrict__ feat, float* __restrict__ nrm) {
  int wid  = (int)((blockIdx.x * (long)blockDim.x + threadIdx.x) >> 5);
  int lane = threadIdx.x & 31;
  if (wid >= Nn) return;
  float4 v = ((const float4*)(feat + (size_t)wid * 128))[lane];
  float ss = v.x * v.x + v.y * v.y + v.z * v.z + v.w * v.w;
  ss = wave_sum(ss);
  if (lane == 0) nrm[wid] = sqrtf(ss);
}

// ---------------- pack A[N,128] (f32) into per-lane WMMA f16 fragments --------------
// Ah[((rt*4 + kki)*32 + lane)*16 + j] = lane's v16h A-fragment for (row-tile rt, k-chunk kki)
// A 16x32 f16 layout: j<8 -> K = kk + half*8 + j ; j>=8 -> K = kk + 16 + half*8 + (j-8)
__global__ void pack_a(const float* __restrict__ A, _Float16* __restrict__ Ah) {
  int wid  = (int)((blockIdx.x * (long)blockDim.x + threadIdx.x) >> 5);
  int lane = threadIdx.x & 31;
  int rowTiles = Nn / 16;
  if (wid >= rowTiles * 4) return;
  int rt = wid >> 2, kki = wid & 3;
  int half = lane >> 4, mn = lane & 15;
  int row = rt * 16 + mn;
  const float* ap = A + (size_t)row * 128 + kki * 32 + half * 8;
  v16h a;
#pragma unroll
  for (int j = 0; j < 8; ++j) {
    a[j]     = (_Float16)ap[j];
    a[8 + j] = (_Float16)ap[16 + j];
  }
  ((v16h*)Ah)[(size_t)wid * 32 + lane] = a;
}

// ---------------- pack W[128,ncols] into per-lane WMMA f16 B fragments --------------
// B 32x16 f16 layout: j<8 -> K = kk + half*8 + j ; j>=8 -> K = kk + 16 + half*8 + (j-8), column = ct*16 + (lane&15)
__global__ void pack_b(const float* __restrict__ W, _Float16* __restrict__ Bh, int ncols) {
  int wid  = (int)((blockIdx.x * (long)blockDim.x + threadIdx.x) >> 5);
  int lane = threadIdx.x & 31;
  int colTiles = (ncols + 15) >> 4;
  if (wid >= colTiles * 4) return;
  int ct = wid >> 2, kki = wid & 3;
  int half = lane >> 4, mn = lane & 15;
  int col = ct * 16 + mn;
  bool valid = col < ncols;
  v16h b;
#pragma unroll
  for (int j = 0; j < 16; ++j) {
    int k = kki * 32 + half * 8 + (j < 8 ? j : 8 + j);
    b[j] = valid ? (_Float16)W[(size_t)k * ncols + col] : (_Float16)0.0f;
  }
  ((v16h*)Bh)[(size_t)wid * 32 + lane] = b;
}

// ---------------- GEMM on packed fragments: one wave per 16x16 tile -----------------
__global__ void gemm_wmma_f16(const _Float16* __restrict__ Ah, const _Float16* __restrict__ Bh,
                              float* __restrict__ out, int ncols) {
  int wid  = (int)((blockIdx.x * (long)blockDim.x + threadIdx.x) >> 5);
  int lane = threadIdx.x & 31;
  int colTiles = (ncols + 15) >> 4;
  int rowTiles = Nn / 16;
  if (wid >= rowTiles * colTiles) return;
  int rt = wid / colTiles, ct = wid % colTiles;
  int half = lane >> 4, mn = lane & 15;
  const v16h* Ap = (const v16h*)Ah + (size_t)rt * 4 * 32 + lane;
  const v16h* Bp = (const v16h*)Bh + (size_t)ct * 4 * 32 + lane;

  v8f acc = {};
#pragma unroll
  for (int kki = 0; kki < 4; ++kki) {
    v16h a = Ap[kki * 32];
    v16h b = Bp[kki * 32];
    acc = __builtin_amdgcn_wmma_f32_16x16x32_f16(false, a, false, b,
                                                 (short)0, acc, false, false);
  }
  int bcol = ct * 16 + mn;
  if (bcol < ncols) {
#pragma unroll
    for (int r = 0; r < 8; ++r) {         // D: element r -> row = half*8 + r
      int m = rt * 16 + half * 8 + r;
      out[(size_t)m * ncols + bcol] = acc[r];
    }
  }
}

// ---------------- edge attention logits (wave per edge) ----------------
__global__ void edge_logits(const float* __restrict__ feat, const float* __restrict__ nrm,
                            const int* __restrict__ esrc, const int* __restrict__ edst,
                            float* __restrict__ la, float* __restrict__ mb) {
  int wid  = (int)((blockIdx.x * (long)blockDim.x + threadIdx.x) >> 5);
  int lane = threadIdx.x & 31;
  if (wid >= ET) return;
  int s_, d_;
  if (wid < Ee) { s_ = esrc[wid]; d_ = edst[wid]; } else { s_ = d_ = wid - Ee; }
  float4 a = ((const float4*)(feat + (size_t)d_ * 128))[lane];
  float4 b = ((const float4*)(feat + (size_t)s_ * 128))[lane];
  float dot = a.x * b.x + a.y * b.y + a.z * b.z + a.w * b.w;
  dot = wave_sum(dot);
  if (lane == 0) {
    float den  = fmaxf(nrm[d_] * nrm[s_], 1e-8f);
    float cosv = dot / den;
    float av   = (cosv < 0.1f) ? 1e-6f : cosv;
    float lav  = logf(av);
    la[wid] = lav;
    atomicMax(mb + d_, lav);
  }
}

// ---------------- exp(la - m[dst]) and segment sum (thread per edge) ----------------
__global__ void edge_exp(const int* __restrict__ edst, float* __restrict__ la,
                         const float* __restrict__ mb, float* __restrict__ sb) {
  int e = blockIdx.x * blockDim.x + threadIdx.x;
  if (e >= ET) return;
  int d_ = (e < Ee) ? edst[e] : e - Ee;
  float v = expf(la[e] - mb[d_]);
  la[e] = v;
  atomicAdd(sb + d_, v);
}

// ---------------- weighted aggregation: agg[dst] += alpha * h[src] (wave per edge) ----
__global__ void edge_aggregate(const int* __restrict__ esrc, const int* __restrict__ edst,
                               const float* __restrict__ la, const float* __restrict__ sb,
                               const float* __restrict__ h, float* __restrict__ agg,
                               int ncols) {
  int wid  = (int)((blockIdx.x * (long)blockDim.x + threadIdx.x) >> 5);
  int lane = threadIdx.x & 31;
  if (wid >= ET) return;
  int s_, d_;
  if (wid < Ee) { s_ = esrc[wid]; d_ = edst[wid]; } else { s_ = d_ = wid - Ee; }
  float alpha = 0.0f;
  if (lane == 0) alpha = la[wid] / sb[d_];
  alpha = __shfl(alpha, 0, 32);
  const float* hr = h + (size_t)s_ * ncols;
  float* orow = agg + (size_t)d_ * ncols;
  for (int c = lane; c < ncols; c += 32)
    atomicAdd(orow + c, alpha * hr[c]);
}

// ---------------- bias + LayerNorm + ReLU, in place (wave per node, 128 feats) -------
__global__ void ln_relu(float* __restrict__ agg, const float* __restrict__ bias,
                        const float* __restrict__ g, const float* __restrict__ bb) {
  int wid  = (int)((blockIdx.x * (long)blockDim.x + threadIdx.x) >> 5);
  int lane = threadIdx.x & 31;
  if (wid >= Nn) return;
  float4* row = (float4*)(agg + (size_t)wid * 128);
  float4 v  = row[lane];
  float4 b4 = ((const float4*)bias)[lane];
  v.x += b4.x; v.y += b4.y; v.z += b4.z; v.w += b4.w;
  float mu = wave_sum(v.x + v.y + v.z + v.w) * (1.0f / 128.0f);
  float dx = v.x - mu, dy = v.y - mu, dz = v.z - mu, dw = v.w - mu;
  float var = wave_sum(dx * dx + dy * dy + dz * dz + dw * dw) * (1.0f / 128.0f);
  float inv = rsqrtf(var + 1e-5f);
  float4 g4 = ((const float4*)g)[lane];
  float4 e4 = ((const float4*)bb)[lane];
  float4 o;
  o.x = fmaxf(dx * inv * g4.x + e4.x, 0.0f);
  o.y = fmaxf(dy * inv * g4.y + e4.y, 0.0f);
  o.z = fmaxf(dz * inv * g4.z + e4.z, 0.0f);
  o.w = fmaxf(dw * inv * g4.w + e4.w, 0.0f);
  row[lane] = o;
}

// ---------------- add bias + row-wise log_softmax over 40 classes (wave per node) ----
__global__ void final_logsoftmax(const float* __restrict__ agg2, const float* __restrict__ b2,
                                 float* __restrict__ out) {
  int wid  = (int)((blockIdx.x * (long)blockDim.x + threadIdx.x) >> 5);
  int lane = threadIdx.x & 31;
  if (wid >= Nn) return;
  const float* row = agg2 + (size_t)wid * 40;
  int c0 = lane, c1 = lane + 32;
  float x0 = row[c0] + b2[c0];                         // c0 < 32 < 40 always valid
  float x1 = (c1 < 40) ? (row[c1] + b2[c1]) : -3.4e38f;
  float mx = wave_max(fmaxf(x0, x1));
  float s  = expf(x0 - mx) + ((c1 < 40) ? expf(x1 - mx) : 0.0f);
  s = wave_sum(s);
  float ls = logf(s);
  float* orow = out + (size_t)wid * 40;
  orow[c0] = x0 - mx - ls;
  if (c1 < 40) orow[c1] = x1 - mx - ls;
}

// ---------------- launcher ----------------
extern "C" void kernel_launch(void* const* d_in, const int* in_sizes, int n_in,
                              void* d_out, int out_size, void* d_ws, size_t ws_size,
                              hipStream_t stream) {
  const float* x    = (const float*)d_in[0];
  const int*   esrc = (const int*)d_in[1];
  const int*   edst = (const int*)d_in[2];
  const float* W1   = (const float*)d_in[3];
  const float* b1   = (const float*)d_in[4];
  const float* lng  = (const float*)d_in[5];
  const float* lnb  = (const float*)d_in[6];
  const float* W2   = (const float*)d_in[7];
  const float* b2   = (const float*)d_in[8];
  float* out = (float*)d_out;

  // workspace layout
  float* ws   = (float*)d_ws;
  float* nrm  = ws;                        // N
  float* mb   = nrm + Nn;                  // N   (segment max)
  float* sb   = mb + Nn;                   // N   (segment sum)
  float* la   = sb + Nn;                   // ET  (edge logits -> exp)
  float* hbuf = la + ET;                   // N*128 (h1, then h2 reuses)
  float* agg  = hbuf + (size_t)Nn * 128;   // N*128 (agg1 -> LN/ReLU in place = hln)
  float* agg2 = agg + (size_t)Nn * 128;    // N*40
  _Float16* Ah = (_Float16*)(agg2 + (size_t)Nn * 40);  // N*128 halves (packed A frags)
  _Float16* Bh = Ah + (size_t)Nn * 128;                // <=16384 halves (packed B frags)

  const int TB = 256;
  auto cdiv = [](long a, long b) { return (int)((a + b - 1) / b); };
  const int rowTiles = Nn / 16;

  // ===== layer 1 =====
  fill_f32<<<cdiv(Nn, TB), TB, 0, stream>>>(mb, Nn, -3.4e38f);
  fill_f32<<<cdiv(Nn, TB), TB, 0, stream>>>(sb, Nn, 0.0f);
  fill_f32<<<cdiv((long)Nn * 128, TB), TB, 0, stream>>>(agg, (long)Nn * 128, 0.0f);
  node_norm<<<cdiv((long)Nn * 32, TB), TB, 0, stream>>>(x, nrm);
  pack_a<<<cdiv((long)rowTiles * 4 * 32, TB), TB, 0, stream>>>(x, Ah);
  pack_b<<<cdiv((long)8 * 4 * 32, TB), TB, 0, stream>>>(W1, Bh, 128);
  gemm_wmma_f16<<<cdiv((long)rowTiles * 8 * 32, 128), 128, 0, stream>>>(Ah, Bh, hbuf, 128);
  edge_logits<<<cdiv((long)ET * 32, TB), TB, 0, stream>>>(x, nrm, esrc, edst, la, mb);
  edge_exp<<<cdiv(ET, TB), TB, 0, stream>>>(edst, la, mb, sb);
  edge_aggregate<<<cdiv((long)ET * 32, TB), TB, 0, stream>>>(esrc, edst, la, sb, hbuf, agg, 128);
  ln_relu<<<cdiv((long)Nn * 32, TB), TB, 0, stream>>>(agg, b1, lng, lnb);

  // ===== layer 2 (features = agg, now LN+ReLU'd) =====
  fill_f32<<<cdiv(Nn, TB), TB, 0, stream>>>(mb, Nn, -3.4e38f);
  fill_f32<<<cdiv(Nn, TB), TB, 0, stream>>>(sb, Nn, 0.0f);
  fill_f32<<<cdiv((long)Nn * 40, TB), TB, 0, stream>>>(agg2, (long)Nn * 40, 0.0f);
  node_norm<<<cdiv((long)Nn * 32, TB), TB, 0, stream>>>(agg, nrm);
  pack_a<<<cdiv((long)rowTiles * 4 * 32, TB), TB, 0, stream>>>(agg, Ah);
  pack_b<<<cdiv((long)3 * 4 * 32, TB), TB, 0, stream>>>(W2, Bh, 40);
  gemm_wmma_f16<<<cdiv((long)rowTiles * 3 * 32, 128), 128, 0, stream>>>(Ah, Bh, hbuf, 40);
  edge_logits<<<cdiv((long)ET * 32, TB), TB, 0, stream>>>(agg, nrm, esrc, edst, la, mb);
  edge_exp<<<cdiv(ET, TB), TB, 0, stream>>>(edst, la, mb, sb);
  edge_aggregate<<<cdiv((long)ET * 32, TB), TB, 0, stream>>>(esrc, edst, la, sb, hbuf, agg2, 40);
  final_logsoftmax<<<cdiv((long)Nn * 32, TB), TB, 0, stream>>>(agg2, b2, out);
}
